// MultiHeadAttention_89300960019273
// MI455X (gfx1250) — matrix operand order repaired and checked by reading, not compiled, not run on hardware
//
#include <hip/hip_runtime.h>
#include <hip/hip_bf16.h>

// ---------------------------------------------------------------------------
// MI455X (gfx1250) fused MHA + residual + LayerNorm, bf16 WMMA pipeline.
// Final: software-pipelined ping-pong GEMMs (loads issued one iteration ahead,
// partial loadcnt waits; residual movs co-execute with the WMMA pipe) +
// async-to-LDS double-buffered flash attention.
// ---------------------------------------------------------------------------

typedef __attribute__((ext_vector_type(16))) __bf16 v16bf;
typedef __attribute__((ext_vector_type(8)))  __bf16 v8bf;
typedef __attribute__((ext_vector_type(8)))  float  v8f;

#define BATCH 2
#define SEQ   2048
#define DIM   1024
#define HEADS 16
#define HDIM  64
#define BS    (BATCH * SEQ)          /* 4096 rows */
#define ATT_SCALE 0.125f             /* 64^-0.5 */
#define LN_EPS 1e-5f

// Workspace layout (bytes)
#define OFF_XH  ((size_t)0)              // bf16 [BS][DIM]           8 MB
#define OFF_WT  ((size_t)8  << 20)       // bf16 [4][DIM][DIM] (N,K) 8 MB
#define OFF_Q   ((size_t)16 << 20)       // bf16 [B*H][SEQ][HDIM]    8 MB
#define OFF_K   ((size_t)24 << 20)       // bf16 [B*H][SEQ][HDIM]    8 MB
#define OFF_VT  ((size_t)32 << 20)       // bf16 [B*H][HDIM][SEQ]    8 MB
#define OFF_AO  ((size_t)40 << 20)       // bf16 [BS][DIM]           8 MB
#define OFF_YR  ((size_t)48 << 20)       // f32  [BS][DIM]          16 MB

__device__ __forceinline__ v8bf ld8(const __bf16* p) {
    return *reinterpret_cast<const v8bf*>(p);
}
// One 32-byte fragment load straight into 8 adjacent VGPRs (32B-aligned!).
__device__ __forceinline__ v16bf ld16(const __bf16* p) {
    return *reinterpret_cast<const v16bf*>(p);
}
__device__ __forceinline__ v16bf cat16(v8bf lo, v8bf hi) {
    v16bf r;
#pragma unroll
    for (int i = 0; i < 8; ++i) { r[i] = lo[i]; r[i + 8] = hi[i]; }
    return r;
}
__device__ __forceinline__ v8f zero8() {
    v8f z;
#pragma unroll
    for (int i = 0; i < 8; ++i) z[i] = 0.0f;
    return z;
}
__device__ __forceinline__ v8f wmma_bf16(v16bf a, v16bf b, v8f c) {
    return __builtin_amdgcn_wmma_f32_16x16x32_bf16(false, a, false, b,
                                                   (short)0, c, false, false);
}
// Async DMA of 16 bytes/lane from global into LDS (gfx1250, ASYNCcnt-tracked).
__device__ __forceinline__ void async_b128(unsigned lds_off, const void* gaddr) {
    asm volatile("global_load_async_to_lds_b128 %0, %1, off"
                 :: "v"(lds_off), "v"((unsigned long long)(size_t)gaddr)
                 : "memory");
}
__device__ __forceinline__ void wait_async0() {
    asm volatile("s_wait_asynccnt 0x0" ::: "memory");
}
// Generic->LDS byte offset: LDS aperture keeps offset in addr[31:0].
__device__ __forceinline__ unsigned lds_off32(const void* p) {
    return (unsigned)(size_t)p;
}

// ---------------------------------------------------------------------------
// 1) x (f32) -> xh (bf16), 8 elements/thread
// ---------------------------------------------------------------------------
__global__ __launch_bounds__(256) void conv_x_kernel(const float* __restrict__ x,
                                                     __bf16* __restrict__ xh) {
    size_t i = ((size_t)blockIdx.x * 256 + threadIdx.x) * 8;
    float4 a = *reinterpret_cast<const float4*>(x + i);
    float4 b = *reinterpret_cast<const float4*>(x + i + 4);
    v8bf r;
    r[0] = (__bf16)a.x; r[1] = (__bf16)a.y; r[2] = (__bf16)a.z; r[3] = (__bf16)a.w;
    r[4] = (__bf16)b.x; r[5] = (__bf16)b.y; r[6] = (__bf16)b.z; r[7] = (__bf16)b.w;
    *reinterpret_cast<v8bf*>(xh + i) = r;
}

// ---------------------------------------------------------------------------
// 2) Transpose + convert the 4 weight matrices: W (K,N) f32 -> Wt (N,K) bf16
// ---------------------------------------------------------------------------
__global__ __launch_bounds__(256) void prep_w_kernel(const float* __restrict__ Wq,
                                                     const float* __restrict__ Wk,
                                                     const float* __restrict__ Wv,
                                                     const float* __restrict__ Wo,
                                                     __bf16* __restrict__ wt) {
    __shared__ float tile[32][33];
    const int wsel = blockIdx.z;
    const float* W = (wsel == 0) ? Wq : (wsel == 1) ? Wk : (wsel == 2) ? Wv : Wo;
    const int tx = threadIdx.x, ty = threadIdx.y;     // (32, 8)
    const int nb = blockIdx.x * 32, kb = blockIdx.y * 32;
#pragma unroll
    for (int i = 0; i < 32; i += 8)
        tile[ty + i][tx] = W[(size_t)(kb + ty + i) * DIM + nb + tx];
    __syncthreads();
    __bf16* o = wt + (size_t)wsel * DIM * DIM;
#pragma unroll
    for (int i = 0; i < 32; i += 8)
        o[(size_t)(nb + ty + i) * DIM + kb + tx] = (__bf16)tile[tx][ty + i];
}

// ---------------------------------------------------------------------------
// 3) Fused QKV GEMM, copy-free 2x-unrolled pipeline. 1 wave = 16x64 tile.
//    grid = (256, 6), block = 256 (8 waves); each wave sits in one weight.
// ---------------------------------------------------------------------------
__global__ __launch_bounds__(256) void qkv_gemm_kernel(const __bf16* __restrict__ xh,
                                                       const __bf16* __restrict__ wt,
                                                       const float* __restrict__ bq,
                                                       const float* __restrict__ bk,
                                                       const float* __restrict__ bv,
                                                       __bf16* __restrict__ qm,
                                                       __bf16* __restrict__ km,
                                                       __bf16* __restrict__ vtm) {
    const int lane = threadIdx.x & 31;
    const int wave = threadIdx.x >> 5;
    const int half = lane >> 4;
    const int l16  = lane & 15;
    const int Mbase = blockIdx.x * 16;
    const int Nbase = blockIdx.y * 512 + wave * 64;   // 0..3071
    const int w = Nbase >> 10;                        // weight select, wave-uniform

    v8f acc[4];
#pragma unroll
    for (int t = 0; t < 4; ++t) acc[t] = zero8();

    const __bf16* arow  = xh + (size_t)(Mbase + l16) * DIM;
    const __bf16* wbase = wt + (size_t)w * DIM * DIM;
    const __bf16* brows[4];
#pragma unroll
    for (int t = 0; t < 4; ++t) {
        const int nrow = (Nbase + t * 16 + l16) & (DIM - 1);
        brows[t] = wbase + (size_t)nrow * DIM;
    }
    auto loadA = [&](int kk) {
        return ld16(arow + kk + half * 16);
    };
    auto loadB = [&](int t, int kk) {       // 32 contiguous bytes, 32B-aligned
        return ld16(brows[t] + kk + half * 16);
    };

    // ping-pong register sets: every set is written only by loads, never moves
    v16bf a0 = loadA(0), a1;
    v16bf b0[4], b1[4];
#pragma unroll
    for (int t = 0; t < 4; ++t) b0[t] = loadB(t, 0);

    for (int kk = 0; kk < DIM - 64; kk += 64) {
        __builtin_prefetch(arow + kk + 256, 0, 2);
        a1 = loadA(kk + 32);
#pragma unroll
        for (int t = 0; t < 4; ++t) b1[t] = loadB(t, kk + 32);
#pragma unroll
        for (int t = 0; t < 4; ++t) acc[t] = wmma_bf16(a0, b0[t], acc[t]);
        a0 = loadA(kk + 64);
#pragma unroll
        for (int t = 0; t < 4; ++t) b0[t] = loadB(t, kk + 64);
#pragma unroll
        for (int t = 0; t < 4; ++t) acc[t] = wmma_bf16(a1, b1[t], acc[t]);
    }
    // tail: chunks DIM-64 (in set 0) and DIM-32
    a1 = loadA(DIM - 32);
#pragma unroll
    for (int t = 0; t < 4; ++t) b1[t] = loadB(t, DIM - 32);
#pragma unroll
    for (int t = 0; t < 4; ++t) acc[t] = wmma_bf16(a0, b0[t], acc[t]);
#pragma unroll
    for (int t = 0; t < 4; ++t) acc[t] = wmma_bf16(a1, b1[t], acc[t]);

    const float* bias = (w == 0) ? bq : (w == 1) ? bk : bv;
#pragma unroll
    for (int t = 0; t < 4; ++t) {
        const int cg = Nbase + t * 16 + l16;
        const int d  = cg & (DIM - 1);
        const int h  = d >> 6, dh = d & 63;
        const float bb = bias[d];
#pragma unroll
        for (int v = 0; v < 8; ++v) {
            const int row = Mbase + half * 8 + v;       // C/D layout row
            const int b_  = row >> 11, s = row & (SEQ - 1);
            const int bh  = b_ * HEADS + h;
            const float val = acc[t][v] + bb;
            if (w == 0)
                qm[((size_t)bh * SEQ + s) * HDIM + dh] = (__bf16)val;
            else if (w == 1)
                km[((size_t)bh * SEQ + s) * HDIM + dh] = (__bf16)val;
            else
                vtm[((size_t)bh * HDIM + dh) * SEQ + s] = (__bf16)val;
        }
    }
}

// ---------------------------------------------------------------------------
// 4) Flash attention with async-to-LDS double-buffered K/V staging.
//    grid = (B*H, SEQ/64), block = 128 (4 waves); each wave = 16 queries.
// ---------------------------------------------------------------------------
__device__ __forceinline__ void stage_kv(const __bf16* km, const __bf16* vtm,
                                         int bh, int kc,
                                         __bf16* kdst, __bf16* vdst, int tid) {
    const char* kg = (const char*)(km + ((size_t)bh * SEQ + kc) * HDIM); // 4KB blob
    const char* vg = (const char*)(vtm + (size_t)bh * HDIM * SEQ + kc);  // 64 rows
    const unsigned kb = lds_off32(kdst);
    const unsigned vb = lds_off32(vdst);
#pragma unroll
    for (int i = 0; i < 2; ++i) {
        const int u = tid * 2 + i;                     // 0..255, 16B units
        async_b128(kb + u * 16, kg + (size_t)u * 16);  // contiguous K tile
        const int row = u >> 2, c = (u & 3) * 16;      // V: 4 units per dh row
        async_b128(vb + u * 16, vg + (size_t)row * (SEQ * 2) + c);
    }
}

__global__ __launch_bounds__(128) void flash_attn_kernel(const __bf16* __restrict__ qm,
                                                         const __bf16* __restrict__ km,
                                                         const __bf16* __restrict__ vtm,
                                                         __bf16* __restrict__ ao) {
    __shared__ __align__(16) __bf16 kbuf[2][32 * 64];   // [key][dh]
    __shared__ __align__(16) __bf16 vbuf[2][64 * 32];   // [dh][key]
    __shared__ __align__(16) __bf16 pbuf[4][16 * 40];   // P tile per wave
    const int tid  = threadIdx.x;
    const int lane = tid & 31;
    const int wave = tid >> 5;
    const int half = lane >> 4;
    const int l16  = lane & 15;
    const int bh   = blockIdx.x;               // = b*HEADS + h
    const int h    = bh & (HEADS - 1);
    const int b_   = bh >> 4;
    const int qbase = blockIdx.y * 64 + wave * 16;

    // Q A-fragments with softmax scale pre-folded (row = qbase + (lane&15))
    const __bf16* qrow = qm + ((size_t)bh * SEQ + qbase + l16) * HDIM;
    v16bf aq0 = ld16(qrow + half * 16);
    v16bf aq1 = ld16(qrow + 32 + half * 16);
#pragma unroll
    for (int i = 0; i < 16; ++i) {
        aq0[i] = (__bf16)((float)aq0[i] * ATT_SCALE);
        aq1[i] = (__bf16)((float)aq1[i] * ATT_SCALE);
    }

    v8f o[4];
#pragma unroll
    for (int t = 0; t < 4; ++t) o[t] = zero8();
    float mrow[8], lrow[8];
#pragma unroll
    for (int v = 0; v < 8; ++v) { mrow[v] = -3.0e38f; lrow[v] = 0.0f; }

    __bf16* myp = &pbuf[wave][0];

    // prologue: DMA chunk 0 into buffer 0
    stage_kv(km, vtm, bh, 0, &kbuf[0][0], &vbuf[0][0], tid);
    wait_async0();
    __syncthreads();

    int buf = 0;
    for (int kc = 0; kc < SEQ; kc += 32) {
        // kick off DMA of the next chunk into the other buffer (overlaps WMMA)
        if (kc + 32 < SEQ)
            stage_kv(km, vtm, bh, kc + 32, &kbuf[buf ^ 1][0], &vbuf[buf ^ 1][0], tid);

        const __bf16* kb = &kbuf[buf][0];
        const __bf16* vb = &vbuf[buf][0];

        // ---- scores S = (Q*scale) Kt for 32 keys (two 16-key tiles)
        v8f st[2];
#pragma unroll
        for (int sub = 0; sub < 2; ++sub) {
            const __bf16* krow = kb + (sub * 16 + l16) * HDIM + half * 16;
            const v16bf bk0 = ld16(krow);        // dh  0..31 slice (32B)
            const v16bf bk1 = ld16(krow + 32);   // dh 32..63 slice (32B)
            v8f s = zero8();
            s = wmma_bf16(aq0, bk0, s);
            s = wmma_bf16(aq1, bk1, s);
            st[sub] = s;
        }
        // ---- online softmax (rows live on 16-lane halves; reduce over cols)
        float alpha[8];
#pragma unroll
        for (int v = 0; v < 8; ++v) {
            float cmax = fmaxf(st[0][v], st[1][v]);
#pragma unroll
            for (int msk = 1; msk < 16; msk <<= 1)
                cmax = fmaxf(cmax, __shfl_xor(cmax, msk, 32));
            const float mnew = fmaxf(mrow[v], cmax);
            alpha[v] = __expf(mrow[v] - mnew);
            mrow[v]  = mnew;
            const float p0 = __expf(st[0][v] - mnew);
            const float p1 = __expf(st[1][v] - mnew);
            st[0][v] = p0; st[1][v] = p1;
            float rs = p0 + p1;
#pragma unroll
            for (int msk = 1; msk < 16; msk <<= 1)
                rs += __shfl_xor(rs, msk, 32);
            lrow[v] = lrow[v] * alpha[v] + rs;
        }
#pragma unroll
        for (int t = 0; t < 4; ++t)
#pragma unroll
            for (int v = 0; v < 8; ++v) o[t][v] *= alpha[v];

        // ---- P (D-layout f32) -> LDS (row-major 16x32 bf16, stride 40)
#pragma unroll
        for (int v = 0; v < 8; ++v) {
            const int r = half * 8 + v;
            myp[r * 40 + l16]      = (__bf16)st[0][v];
            myp[r * 40 + 16 + l16] = (__bf16)st[1][v];
        }
        asm volatile("s_wait_dscnt 0x0" ::: "memory");   // DS wave-ordering fence
        const v16bf ap = cat16(ld8(myp + l16 * 40 + half * 8),
                               ld8(myp + l16 * 40 + 16 + half * 8));

        // ---- O += P x V  (V tile in LDS, (dh,key) rows)
#pragma unroll
        for (int t = 0; t < 4; ++t) {
            const v16bf bv = cat16(ld8(vb + (t * 16 + l16) * 32 + half * 8),
                                   ld8(vb + (t * 16 + l16) * 32 + 16 + half * 8));
            o[t] = wmma_bf16(ap, bv, o[t]);
        }

        // next chunk's DMA must be complete and everyone done reading this one
        wait_async0();
        __syncthreads();
        buf ^= 1;
    }

    // ---- normalize and scatter into attention-output (B,S,D) bf16
#pragma unroll
    for (int v = 0; v < 8; ++v) lrow[v] = 1.0f / lrow[v];
#pragma unroll
    for (int t = 0; t < 4; ++t)
#pragma unroll
        for (int v = 0; v < 8; ++v) {
            const int row = qbase + half * 8 + v;
            const size_t gr = (size_t)(b_ * SEQ + row);
            ao[gr * DIM + h * HDIM + t * 16 + l16] = (__bf16)(o[t][v] * lrow[v]);
        }
}

// ---------------------------------------------------------------------------
// 5) Output projection + bias + residual (copy-free 2x-unrolled pipeline).
//    grid = (256, 2), block = 256.
// ---------------------------------------------------------------------------
__global__ __launch_bounds__(256) void out_gemm_kernel(const __bf16* __restrict__ ao,
                                                       const __bf16* __restrict__ wt,
                                                       const float* __restrict__ bo,
                                                       const float* __restrict__ x,
                                                       float* __restrict__ yr) {
    const int lane = threadIdx.x & 31;
    const int wave = threadIdx.x >> 5;
    const int half = lane >> 4;
    const int l16  = lane & 15;
    const int Mbase = blockIdx.x * 16;
    const int Nbase = blockIdx.y * 512 + wave * 64;   // 0..1023

    v8f acc[4];
#pragma unroll
    for (int t = 0; t < 4; ++t) acc[t] = zero8();

    const __bf16* arow  = ao + (size_t)(Mbase + l16) * DIM;
    const __bf16* wbase = wt + (size_t)3 * DIM * DIM;   // Wo transposed
    const __bf16* brows[4];
#pragma unroll
    for (int t = 0; t < 4; ++t)
        brows[t] = wbase + (size_t)(Nbase + t * 16 + l16) * DIM;

    auto loadA = [&](int kk) {
        return ld16(arow + kk + half * 16);
    };
    auto loadB = [&](int t, int kk) {       // 32 contiguous bytes, 32B-aligned
        return ld16(brows[t] + kk + half * 16);
    };

    v16bf a0 = loadA(0), a1;
    v16bf b0[4], b1[4];
#pragma unroll
    for (int t = 0; t < 4; ++t) b0[t] = loadB(t, 0);

    for (int kk = 0; kk < DIM - 64; kk += 64) {
        __builtin_prefetch(arow + kk + 256, 0, 2);
        a1 = loadA(kk + 32);
#pragma unroll
        for (int t = 0; t < 4; ++t) b1[t] = loadB(t, kk + 32);
#pragma unroll
        for (int t = 0; t < 4; ++t) acc[t] = wmma_bf16(a0, b0[t], acc[t]);
        a0 = loadA(kk + 64);
#pragma unroll
        for (int t = 0; t < 4; ++t) b0[t] = loadB(t, kk + 64);
#pragma unroll
        for (int t = 0; t < 4; ++t) acc[t] = wmma_bf16(a1, b1[t], acc[t]);
    }
    a1 = loadA(DIM - 32);
#pragma unroll
    for (int t = 0; t < 4; ++t) b1[t] = loadB(t, DIM - 32);
#pragma unroll
    for (int t = 0; t < 4; ++t) acc[t] = wmma_bf16(a0, b0[t], acc[t]);
#pragma unroll
    for (int t = 0; t < 4; ++t) acc[t] = wmma_bf16(a1, b1[t], acc[t]);

#pragma unroll
    for (int t = 0; t < 4; ++t) {
        const int c = Nbase + t * 16 + l16;
        const float bb = bo[c];
#pragma unroll
        for (int v = 0; v < 8; ++v) {
            const int row = Mbase + half * 8 + v;
            const size_t idx = (size_t)row * DIM + c;
            yr[idx] = acc[t][v] + bb + x[idx];
        }
    }
}

// ---------------------------------------------------------------------------
// 6) LayerNorm over D=1024 per row. block = 256 (8 waves), 4 elems/thread.
// ---------------------------------------------------------------------------
__global__ __launch_bounds__(256) void layer_norm_kernel(const float* __restrict__ yr,
                                                         const float* __restrict__ gamma,
                                                         const float* __restrict__ beta,
                                                         float* __restrict__ out) {
    __shared__ float red[2][8];
    const int row = blockIdx.x;
    const int base = threadIdx.x * 4;
    const float* r = yr + (size_t)row * DIM;
    const float4 v4 = *reinterpret_cast<const float4*>(r + base);
    float s  = v4.x + v4.y + v4.z + v4.w;
    float s2 = v4.x * v4.x + v4.y * v4.y + v4.z * v4.z + v4.w * v4.w;
#pragma unroll
    for (int msk = 1; msk < 32; msk <<= 1) {
        s  += __shfl_xor(s,  msk, 32);
        s2 += __shfl_xor(s2, msk, 32);
    }
    const int wave = threadIdx.x >> 5, lane = threadIdx.x & 31;
    if (lane == 0) { red[0][wave] = s; red[1][wave] = s2; }
    __syncthreads();
    if (wave == 0) {
        float a = (lane < 8) ? red[0][lane] : 0.0f;
        float b = (lane < 8) ? red[1][lane] : 0.0f;
#pragma unroll
        for (int msk = 1; msk < 8; msk <<= 1) {
            a += __shfl_xor(a, msk, 32);
            b += __shfl_xor(b, msk, 32);
        }
        if (lane == 0) { red[0][0] = a; red[1][0] = b; }
    }
    __syncthreads();
    const float mu  = red[0][0] * (1.0f / DIM);
    const float var = red[1][0] * (1.0f / DIM) - mu * mu;
    const float rs  = rsqrtf(var + LN_EPS);
    float* op = out + (size_t)row * DIM + base;
    op[0] = (v4.x - mu) * rs * gamma[base + 0] + beta[base + 0];
    op[1] = (v4.y - mu) * rs * gamma[base + 1] + beta[base + 1];
    op[2] = (v4.z - mu) * rs * gamma[base + 2] + beta[base + 2];
    op[3] = (v4.w - mu) * rs * gamma[base + 3] + beta[base + 3];
}

// ---------------------------------------------------------------------------
extern "C" void kernel_launch(void* const* d_in, const int* in_sizes, int n_in,
                              void* d_out, int out_size, void* d_ws, size_t ws_size,
                              hipStream_t stream) {
    (void)in_sizes; (void)n_in; (void)out_size; (void)ws_size;
    const float* x     = (const float*)d_in[0];
    const float* Wq    = (const float*)d_in[1];
    const float* bq    = (const float*)d_in[2];
    const float* Wk    = (const float*)d_in[3];
    const float* bk    = (const float*)d_in[4];
    const float* Wv    = (const float*)d_in[5];
    const float* bv    = (const float*)d_in[6];
    const float* Wo    = (const float*)d_in[7];
    const float* bo    = (const float*)d_in[8];
    const float* gamma = (const float*)d_in[9];
    const float* beta  = (const float*)d_in[10];

    char* ws = (char*)d_ws;
    __bf16* xh  = (__bf16*)(ws + OFF_XH);
    __bf16* wt  = (__bf16*)(ws + OFF_WT);
    __bf16* qm  = (__bf16*)(ws + OFF_Q);
    __bf16* km  = (__bf16*)(ws + OFF_K);
    __bf16* vtm = (__bf16*)(ws + OFF_VT);
    __bf16* ao  = (__bf16*)(ws + OFF_AO);
    float*  yr  = (float*)(ws + OFF_YR);
    float*  out = (float*)d_out;

    conv_x_kernel<<<(BS * DIM) / (256 * 8), 256, 0, stream>>>(x, xh);
    prep_w_kernel<<<dim3(DIM / 32, DIM / 32, 4), dim3(32, 8), 0, stream>>>(Wq, Wk, Wv, Wo, wt);
    qkv_gemm_kernel<<<dim3(BS / 16, 6), 256, 0, stream>>>(xh, wt, bq, bk, bv, qm, km, vtm);
    flash_attn_kernel<<<dim3(BATCH * HEADS, SEQ / 64), 128, 0, stream>>>(qm, km, vtm, ao);
    out_gemm_kernel<<<dim3(BS / 16, 2), 256, 0, stream>>>(ao, wt, bo, x, yr);
    layer_norm_kernel<<<BS, 256, 0, stream>>>(yr, gamma, beta, out);
}
